// BasicMultiHeadSelfAttention_18811956757272
// MI455X (gfx1250) — matrix operand order repaired and checked by reading, not compiled, not run on hardware
//
#include <hip/hip_runtime.h>
#include <hip/hip_bf16.h>

// ---------------------------------------------------------------------------
// Causal multi-head self-attention, B=2 S=2048 E=1024 H=16 D=64, on gfx1250.
// f16 WMMA (v_wmma_f32_16x16x32_f16) for all matmul stages, fp32 accumulate.
// GEMMs register-blocked 64x32 per wave (8 WMMA / 6 fragment loads per K-step
// -> ~21 FLOP/byte from L0, WMMA-limited instead of load-limited).
// Attention K-tiles staged to LDS by the Tensor Data Mover (double-buffered,
// TENSORcnt-synchronized) when the TDM builtin is available.
// ---------------------------------------------------------------------------

constexpr int cB = 2;
constexpr int cS = 2048;
constexpr int cE = 1024;
constexpr int cH = 16;
constexpr int cD = 64;           // head dim
constexpr int cBS = cB * cS;     // 4096 rows
constexpr int c3E = 3 * cE;      // 3072

typedef _Float16 half8  __attribute__((ext_vector_type(8)));
typedef _Float16 half16 __attribute__((ext_vector_type(16)));
typedef float    float8 __attribute__((ext_vector_type(8)));
typedef unsigned int uint4v __attribute__((ext_vector_type(4)));
typedef int      int4v  __attribute__((ext_vector_type(4)));
typedef int      int8v  __attribute__((ext_vector_type(8)));

#if defined(__HIP_DEVICE_COMPILE__) && __has_builtin(__builtin_amdgcn_tensor_load_to_lds)
#define HAVE_TDM 1
#else
#define HAVE_TDM 0
#endif

#if HAVE_TDM
#define TDM_WAIT(n) __builtin_amdgcn_s_wait_tensorcnt((short)(n))
#else
#define TDM_WAIT(n)
#endif

static __device__ __forceinline__ float8 wmma16x16x32(half16 a, half16 b, float8 c) {
  // (neg_a, A, neg_b, B, c_mod, C, reuse_a, reuse_b)
  return __builtin_amdgcn_wmma_f32_16x16x32_f16(false, a, false, b, (short)0, c,
                                                false, false);
}

// A-fragment (16x32 f16), row-major source, leading dim `ld` in halves.
// ISA 7.12.2: lanes 0-15 row=lane, K in {k..k+7, k+16..k+23};
//             lanes 16-31 row=lane-16, K in {k+8..k+15, k+24..k+31}.
static __device__ __forceinline__ half16 frag_a(const _Float16* __restrict__ base,
                                                int row, int ld, int k, int lane) {
  const _Float16* p = base + (size_t)row * ld + k + ((lane & 16) ? 8 : 0);
  union { half16 v; half8 h[2]; } u;
  u.h[0] = *(const half8*)(p);
  u.h[1] = *(const half8*)(p + 16);
  return u.v;
}

// B-fragment (32x16 f16) from a transposed matrix BT[n][k] (ld halves).
// Lanes 0-15: n=lane, K in {k..k+7, k+16..k+23}; lanes 16-31: n=lane-16, K in {k+8..k+15, k+24..k+31}.
static __device__ __forceinline__ half16 frag_b(const _Float16* __restrict__ baseT,
                                                int n, int ld, int k, int lane) {
  const _Float16* p = baseT + (size_t)n * ld + k + ((lane & 16) ? 8 : 0);
  union { half16 v; half8 h[2]; } u;
  u.h[0] = *(const half8*)(p);
  u.h[1] = *(const half8*)(p + 16);
  return u.v;
}

// Tensor Data Mover: DMA a 2D f16 tile (tile0 x tile1 elements, row stride
// `stride0` elements) from global `gaddr` into LDS byte offset `lds_off`.
// D# bit layout per CDNA5 ISA ch.8 (group0: count/lds/global/type=2;
// group1: data_size=2B, dims, tile dims, dim0 stride).
static __device__ __forceinline__ void tdm_load_2d(const void* gaddr, unsigned lds_off,
                                                   unsigned dim0, unsigned dim1,
                                                   unsigned tile0, unsigned tile1,
                                                   unsigned stride0) {
#if HAVE_TDM
  unsigned long long ga = (unsigned long long)(uintptr_t)gaddr;
  uint4v g0;
  g0[0] = 1u;                                            // count=1 (valid user D#)
  g0[1] = lds_off;                                       // lds_addr [63:32]
  g0[2] = (unsigned)ga;                                  // global_addr [95:64]
  g0[3] = (unsigned)((ga >> 32) & 0x01FFFFFFu) | (2u << 30); // addr[56:32]|type=2
  int8v g1;
  g1[0] = (int)(1u << 16);                               // data_size=1 -> 2 bytes
  g1[1] = (int)((dim0 & 0xFFFFu) << 16);                 // tensor_dim0 lo16
  g1[2] = (int)((dim0 >> 16) | ((dim1 & 0xFFFFu) << 16));// dim0 hi | dim1 lo
  g1[3] = (int)((dim1 >> 16) | (tile0 << 16));           // dim1 hi | tile_dim0
  g1[4] = (int)(tile1 & 0xFFFFu);                        // tile_dim1 (tile_dim2=0)
  g1[5] = (int)stride0;                                  // tensor_dim0_stride lo32
  g1[6] = 0;                                             // stride0 hi | stride1 lo
  g1[7] = 0;                                             // stride1 hi
  int4v z4 = {};
#if __clang_major__ >= 23
  int8v z8 = {};
  __builtin_amdgcn_tensor_load_to_lds(g0, g1, z4, z4, z8, 0);
#else
  __builtin_amdgcn_tensor_load_to_lds(g0, g1, z4, z4, 0);
#endif
#else
  (void)gaddr; (void)lds_off; (void)dim0; (void)dim1;
  (void)tile0; (void)tile1; (void)stride0;
#endif
}

// ---------------------------------------------------------------------------
// Precision-conversion / layout prep (bandwidth-trivial: < 30 MB total)
// ---------------------------------------------------------------------------
__global__ void k_f32_to_f16(const float* __restrict__ src,
                             _Float16* __restrict__ dst, int n) {
  int i = blockIdx.x * blockDim.x + threadIdx.x;
  if (i < n) dst[i] = (_Float16)src[i];
}

// W[K][N] (f32) -> WT[N][K] (f16); write side coalesced.
__global__ void k_transpose_f16(const float* __restrict__ W,
                                _Float16* __restrict__ WT, int K, int N) {
  int i = blockIdx.x * blockDim.x + threadIdx.x;
  if (i < K * N) {
    int n = i / K;
    int k = i - n * K;
    WT[i] = (_Float16)W[(size_t)k * N + n];
  }
}

// ---------------------------------------------------------------------------
// GEMM1: qkv = Xh[BS,E] @ Wqkv + bqkv, scattered into per-head Q/K/Vt (f16).
// 64x32 tile per wave: 4 A-frags + 2 B-frags feed 8 WMMAs per K=32 step.
// ---------------------------------------------------------------------------
__global__ void __launch_bounds__(256)
k_gemm_qkv(const _Float16* __restrict__ Xh, const _Float16* __restrict__ WT,
           const float* __restrict__ bias, _Float16* __restrict__ Qb,
           _Float16* __restrict__ Kb, _Float16* __restrict__ Vt) {
  const int lane = threadIdx.x & 31;
  const int wid = threadIdx.x >> 5;
  const int tile = blockIdx.x * 8 + wid;
  const int TN = c3E / 32;                    // 96 column blocks
  const int tm = tile / TN, tn = tile % TN;
  const int rowb = tm * 64, colb = tn * 32;
  const int lcol = lane & 15;
  const int lrow = (lane & 16) ? 8 : 0;

  float8 acc[4][2] = {};
  for (int k = 0; k < cE; k += 32) {
    if (k + 32 < cE) {
      __builtin_prefetch(Xh + (size_t)(rowb + lcol) * cE + k + 32, 0, 1);
      __builtin_prefetch(WT + (size_t)(colb + lcol) * cE + k + 32, 0, 1);
    }
    half16 bf0 = frag_b(WT, colb + lcol, cE, k, lane);
    half16 bf1 = frag_b(WT, colb + 16 + lcol, cE, k, lane);
#pragma unroll
    for (int i = 0; i < 4; i++) {
      half16 af = frag_a(Xh, rowb + i * 16 + lcol, cE, k, lane);
      acc[i][0] = wmma16x16x32(af, bf0, acc[i][0]);
      acc[i][1] = wmma16x16x32(af, bf1, acc[i][1]);
    }
  }

#pragma unroll
  for (int j = 0; j < 2; j++) {
    const int col = colb + j * 16 + lcol;
    const float bv = bias[col];
#pragma unroll
    for (int i = 0; i < 4; i++) {
#pragma unroll
      for (int v = 0; v < 8; v++) {
        const int row = rowb + i * 16 + v + lrow;   // row in [0, BS)
        const int b_ = row >> 11;
        const int s_ = row & (cS - 1);
        const float val = acc[i][j][v] + bv;
        if (col < cE) {                             // Q -> [B,H,S,D]
          const int h_ = col >> 6, d_ = col & 63;
          Qb[(((size_t)b_ * cH + h_) * cS + s_) * cD + d_] = (_Float16)val;
        } else if (col < 2 * cE) {                  // K -> [B,H,S,D]
          const int c = col - cE, h_ = c >> 6, d_ = c & 63;
          Kb[(((size_t)b_ * cH + h_) * cS + s_) * cD + d_] = (_Float16)val;
        } else {                                    // V -> transposed [B,H,D,S]
          const int c = col - 2 * cE, h_ = c >> 6, d_ = c & 63;
          Vt[(((size_t)b_ * cH + h_) * cD + d_) * cS + s_] = (_Float16)val;
        }
      }
    }
  }
}

// ---------------------------------------------------------------------------
// Flash attention: one wave per (b,h,16-row q tile), 32 keys per step.
// K tiles (32x64 f16 = 4 KB) DMA'd to LDS by the TDM, double-buffered and
// overlapped with WMMA; V^T rows give contiguous B-fragments from global.
// ---------------------------------------------------------------------------
__global__ void __launch_bounds__(256)
k_attn(const _Float16* __restrict__ Qb, const _Float16* __restrict__ Kb,
       const _Float16* __restrict__ Vt, _Float16* __restrict__ Oh) {
  __shared__ __align__(16) _Float16 Pl[8][16 * 40];    // P staging, padded rows
#if HAVE_TDM
  __shared__ __align__(16) _Float16 Ktile[8][2][32 * cD]; // per-wave dbl buffer
#endif

  const int lane = threadIdx.x & 31;
  const int wid = threadIdx.x >> 5;
  const int tile = blockIdx.x * 8 + wid;
  const int QT = cS / 16;                     // 128 q tiles per (b,h)
  if (tile >= cB * cH * QT) return;           // wave-uniform
  const int qt = tile % QT;
  const int bh = tile / QT;
  const int qb = qt * 16;
  const int bi = bh / cH, hi_ = bh % cH;

  const _Float16* Qp = Qb + (size_t)bh * cS * cD;
  const _Float16* Kp = Kb + (size_t)bh * cS * cD;
  const _Float16* Vp = Vt + (size_t)bh * cD * cS;

  const int lcol = lane & 15;
  const int lrow = (lane & 16) ? 8 : 0;

  // Q fragments for the whole tile (D = 64 -> two K=32 chunks), kept live.
  const half16 q0 = frag_a(Qp, qb + lcol, cD, 0, lane);
  const half16 q1 = frag_a(Qp, qb + lcol, cD, 32, lane);

  float m[8], l[8];
  float8 o0 = {}, o1 = {}, o2 = {}, o3 = {};
#pragma unroll
  for (int v = 0; v < 8; v++) { m[v] = -1e30f; l[v] = 0.f; }

  _Float16* Pw = &Pl[wid][0];

  const int nsteps = (qb + 16 + 31) >> 5;     // causal bound: skip upper tri
#if HAVE_TDM
  tdm_load_2d(Kp, (unsigned)(uintptr_t)(void*)&Ktile[wid][0][0],
              cD, cS, cD, 32, cD);            // prologue: tile for kc = 0
#endif

  for (int s = 0; s < nsteps; s++) {
    const int kc = s << 5;
    const int cur = s & 1;
#if HAVE_TDM
    if (s + 1 < nsteps) {                     // issue next tile, then wait cur
      tdm_load_2d(Kp + (size_t)(kc + 32) * cD,
                  (unsigned)(uintptr_t)(void*)&Ktile[wid][cur ^ 1][0],
                  cD, cS, cD, 32, cD);
      TDM_WAIT(1);
    } else {
      TDM_WAIT(0);
    }
    const _Float16* Kcur = &Ktile[wid][cur][0];   // rows = key-kc, ld = cD
#else
    const _Float16* Kcur = Kp + (size_t)kc * cD;
    (void)cur;
#endif

    // scores: two 16x16 tiles (keys kc..kc+15 and kc+16..kc+31)
    float8 s0 = {}, s1 = {};
    s0 = wmma16x16x32(q0, frag_b(Kcur, lcol, cD, 0, lane), s0);
    s0 = wmma16x16x32(q1, frag_b(Kcur, lcol, cD, 32, lane), s0);
    s1 = wmma16x16x32(q0, frag_b(Kcur, 16 + lcol, cD, 0, lane), s1);
    s1 = wmma16x16x32(q1, frag_b(Kcur, 16 + lcol, cD, 32, lane), s1);

#pragma unroll
    for (int v = 0; v < 8; v++) {
      const int row = qb + v + lrow;
      float x0 = s0[v] * 0.125f;               // 1/sqrt(64)
      float x1 = s1[v] * 0.125f;
      if (kc + lcol > row)      x0 = -1e30f;   // causal mask
      if (kc + 16 + lcol > row) x1 = -1e30f;
      float tm = fmaxf(x0, x1);
#pragma unroll
      for (int off = 1; off < 16; off <<= 1)   // row max across 16 lanes
        tm = fmaxf(tm, __shfl_xor(tm, off, 16));
      const float mn = fmaxf(m[v], tm);
      const float alpha = __expf(m[v] - mn);
      const float p0 = __expf(x0 - mn);
      const float p1 = __expf(x1 - mn);
      float rs = p0 + p1;
#pragma unroll
      for (int off = 1; off < 16; off <<= 1)   // row sum across 16 lanes
        rs += __shfl_xor(rs, off, 16);
      m[v] = mn;
      l[v] = l[v] * alpha + rs;
      o0[v] *= alpha; o1[v] *= alpha; o2[v] *= alpha; o3[v] *= alpha;
      // C-layout -> LDS (row = v+lrow, col = lcol / lcol+16), f16
      Pw[(v + lrow) * 40 + lcol]      = (_Float16)p0;
      Pw[(v + lrow) * 40 + 16 + lcol] = (_Float16)p1;
    }

    // P (16x32) as A-fragment from LDS; V^T rows -> contiguous B-fragments.
    const half16 pa = frag_a(Pw, lcol, 40, 0, lane);
    o0 = wmma16x16x32(pa, frag_b(Vp, 0  + lcol, cS, kc, lane), o0);
    o1 = wmma16x16x32(pa, frag_b(Vp, 16 + lcol, cS, kc, lane), o1);
    o2 = wmma16x16x32(pa, frag_b(Vp, 32 + lcol, cS, kc, lane), o2);
    o3 = wmma16x16x32(pa, frag_b(Vp, 48 + lcol, cS, kc, lane), o3);
  }

#pragma unroll
  for (int v = 0; v < 8; v++) {
    const float inv = 1.f / l[v];
    const int srow = qb + v + lrow;
    const size_t obase = (((size_t)bi * cS + srow) * cH + hi_) * cD + lcol;
    Oh[obase + 0]  = (_Float16)(o0[v] * inv);
    Oh[obase + 16] = (_Float16)(o1[v] * inv);
    Oh[obase + 32] = (_Float16)(o2[v] * inv);
    Oh[obase + 48] = (_Float16)(o3[v] * inv);
  }
}

// ---------------------------------------------------------------------------
// GEMM2: out = Oh[BS,E] @ Wout + bout (fp32 output), 64x32 tile per wave.
// ---------------------------------------------------------------------------
__global__ void __launch_bounds__(256)
k_gemm_out(const _Float16* __restrict__ Oh, const _Float16* __restrict__ WT,
           const float* __restrict__ bias, float* __restrict__ out) {
  const int lane = threadIdx.x & 31;
  const int wid = threadIdx.x >> 5;
  const int tile = blockIdx.x * 8 + wid;
  const int TN = cE / 32;                     // 32 column blocks
  const int tm = tile / TN, tn = tile % TN;
  const int rowb = tm * 64, colb = tn * 32;
  const int lcol = lane & 15;
  const int lrow = (lane & 16) ? 8 : 0;

  float8 acc[4][2] = {};
  for (int k = 0; k < cE; k += 32) {
    if (k + 32 < cE) {
      __builtin_prefetch(Oh + (size_t)(rowb + lcol) * cE + k + 32, 0, 1);
      __builtin_prefetch(WT + (size_t)(colb + lcol) * cE + k + 32, 0, 1);
    }
    half16 bf0 = frag_b(WT, colb + lcol, cE, k, lane);
    half16 bf1 = frag_b(WT, colb + 16 + lcol, cE, k, lane);
#pragma unroll
    for (int i = 0; i < 4; i++) {
      half16 af = frag_a(Oh, rowb + i * 16 + lcol, cE, k, lane);
      acc[i][0] = wmma16x16x32(af, bf0, acc[i][0]);
      acc[i][1] = wmma16x16x32(af, bf1, acc[i][1]);
    }
  }

#pragma unroll
  for (int j = 0; j < 2; j++) {
    const int col = colb + j * 16 + lcol;
    const float bv = bias[col];
#pragma unroll
    for (int i = 0; i < 4; i++) {
#pragma unroll
      for (int v = 0; v < 8; v++) {
        const int row = rowb + i * 16 + v + lrow;
        out[(size_t)row * cE + col] = acc[i][j][v] + bv;
      }
    }
  }
}

// ---------------------------------------------------------------------------
// Host-side orchestration
// ---------------------------------------------------------------------------
extern "C" void kernel_launch(void* const* d_in, const int* in_sizes, int n_in,
                              void* d_out, int out_size, void* d_ws, size_t ws_size,
                              hipStream_t stream) {
  const float* Qin  = (const float*)d_in[0];  // [B,S,E]
  const float* Wqkv = (const float*)d_in[1];  // [E,3E]
  const float* bqkv = (const float*)d_in[2];  // [3E]
  const float* Wout = (const float*)d_in[3];  // [E,E]
  const float* bout = (const float*)d_in[4];  // [E]
  float* out = (float*)d_out;                 // [B,S,E] fp32

  char* ws = (char*)d_ws;
  size_t off = 0;
  auto take = [&](size_t bytes) {
    void* p = ws + off;
    off += (bytes + 255) & ~(size_t)255;
    return p;
  };
  _Float16* Xh    = (_Float16*)take((size_t)cBS * cE * 2);          // 8 MB
  _Float16* WqkvT = (_Float16*)take((size_t)c3E * cE * 2);          // 6 MB
  _Float16* WoutT = (_Float16*)take((size_t)cE * cE * 2);           // 2 MB
  _Float16* Qb    = (_Float16*)take((size_t)cB * cH * cS * cD * 2); // 8 MB
  _Float16* Kb    = (_Float16*)take((size_t)cB * cH * cS * cD * 2); // 8 MB
  _Float16* Vt    = (_Float16*)take((size_t)cB * cH * cD * cS * 2); // 8 MB
  _Float16* Oh    = (_Float16*)take((size_t)cBS * cE * 2);          // 8 MB
  (void)ws_size; (void)in_sizes; (void)n_in; (void)out_size;

  // 1) precision/layout prep
  {
    const int n = cBS * cE;
    k_f32_to_f16<<<(n + 255) / 256, 256, 0, stream>>>(Qin, Xh, n);
  }
  {
    const int n = cE * c3E;
    k_transpose_f16<<<(n + 255) / 256, 256, 0, stream>>>(Wqkv, WqkvT, cE, c3E);
  }
  {
    const int n = cE * cE;
    k_transpose_f16<<<(n + 255) / 256, 256, 0, stream>>>(Wout, WoutT, cE, cE);
  }

  // 2) QKV projection: (BS/64)*(3E/32) = 6144 wave-tiles, 8 waves/block
  k_gemm_qkv<<<(cBS / 64) * (c3E / 32) / 8, 256, 0, stream>>>(Xh, WqkvT, bqkv,
                                                              Qb, Kb, Vt);

  // 3) causal flash attention: B*H*(S/16) = 4096 wave-tiles
  k_attn<<<cB * cH * (cS / 16) / 8, 256, 0, stream>>>(Qb, Kb, Vt, Oh);

  // 4) output projection: (BS/64)*(E/32) = 2048 wave-tiles
  k_gemm_out<<<(cBS / 64) * (cE / 32) / 8, 256, 0, stream>>>(Oh, WoutT, bout, out);
}
